// TriangleMultiplicationOutgoing_52183852646479
// MI455X (gfx1250) — matrix operand order repaired and checked by reading, not compiled, not run on hardware
//
#include <hip/hip_runtime.h>

#define NN 384
#define CC 128
#define MTOT (NN * NN)   // 147456

typedef __attribute__((ext_vector_type(16))) __bf16 v16bf;
typedef __attribute__((ext_vector_type(8)))  float  v8f;

union FragBF { uint4 q[2]; v16bf v; };

__device__ __forceinline__ unsigned short f2bf(float f) {
    unsigned u = __float_as_uint(f);
    u += 0x7FFFu + ((u >> 16) & 1u);          // round-to-nearest-even
    return (unsigned short)(u >> 16);
}
__device__ __forceinline__ float bf2f(unsigned short h) {
    return __uint_as_float(((unsigned)h) << 16);
}
__device__ __forceinline__ float sigmoidf(float x) {
    return 1.0f / (1.0f + __expf(-x));
}
__device__ __forceinline__ float wred(float v) {
#pragma unroll
    for (int off = 16; off >= 1; off >>= 1) v += __shfl_xor(v, off, 32);
    return v;
}

// ---- CDNA5 async copy global->LDS (ASYNCcnt path), with safe fallback ------
#if defined(__has_builtin)
#if __has_builtin(__builtin_amdgcn_global_load_async_to_lds_b128)
#define HAVE_ASYNC_LDS 1
#endif
#endif

typedef int v4i_t __attribute__((vector_size(16)));
typedef __attribute__((address_space(1))) v4i_t* as1_v4i;
typedef __attribute__((address_space(3))) v4i_t* as3_v4i;

__device__ __forceinline__ void async_copy16(const unsigned short* g, unsigned short* l) {
#ifdef HAVE_ASYNC_LDS
    __builtin_amdgcn_global_load_async_to_lds_b128(
        (as1_v4i)(v4i_t*)g, (as3_v4i)(v4i_t*)l, 0, 0);
#else
    *(uint4*)l = *(const uint4*)g;
#endif
}
__device__ __forceinline__ void async_wait0() {
#ifdef HAVE_ASYNC_LDS
#if defined(__has_builtin) && __has_builtin(__builtin_amdgcn_s_wait_asynccnt)
    __builtin_amdgcn_s_wait_asynccnt(0);
#else
    asm volatile("s_wait_asynccnt 0" ::: "memory");
#endif
#endif
}

// A-matrix 16x32 bf16 fragment (ISA 7.12.2): lanes 0-15 hold K {0..7,16..23},
// lanes 16-31 hold K {8..15,24..31} of row (lane&15). rowp = base of this row.
__device__ __forceinline__ v16bf load_afrag(const unsigned short* rowp, int k0, int hi) {
    FragBF f;
    f.q[0] = *(const uint4*)(rowp + k0 + hi * 8);
    f.q[1] = *(const uint4*)(rowp + k0 + 16 + hi * 8);
    return f.v;
}
// B-matrix 32x16 bf16 fragment: lane n=lane&15 holds K {8h..8h+7, 16+8h..16+8h+7}
// (h = lane>>4), matching the A layout; rowp = base of the n-th row of the (n,k) matrix.
__device__ __forceinline__ v16bf load_bfrag(const unsigned short* rowp, int k0, int hi) {
    FragBF f;
    const unsigned short* p = rowp + k0 + hi * 8;
    f.q[0] = *(const uint4*)(p);
    f.q[1] = *(const uint4*)(p + 16);
    return f.v;
}
__device__ __forceinline__ uint4 pack8(const unsigned short* h) {
    uint4 p;
    p.x = (unsigned)h[0] | ((unsigned)h[1] << 16);
    p.y = (unsigned)h[2] | ((unsigned)h[3] << 16);
    p.z = (unsigned)h[4] | ((unsigned)h[5] << 16);
    p.w = (unsigned)h[6] | ((unsigned)h[7] << 16);
    return p;
}

#define WMMA_BF16(A, B, C) \
    __builtin_amdgcn_wmma_f32_16x16x32_bf16(false, (A), false, (B), (short)0, (C), false, false)

// ---------------------------------------------------------------------------
// Weight prep: Wt[n][k] = bf16(W[k][n])  (128x128)
// ---------------------------------------------------------------------------
__global__ void k_wprep(const float* __restrict__ W, unsigned short* __restrict__ Wt) {
    const int n = blockIdx.x, k = threadIdx.x;
    Wt[(size_t)n * CC + k] = f2bf(W[(size_t)k * CC + n]);
}

// ---------------------------------------------------------------------------
// LN1 + 5 projections (GLU for a/b, sigmoid for gate).
// Block = 8 waves; each wave owns 16 rows of zn staged bf16 in LDS.
//  a_t / b_t written channel-major (c, m): transposed D-fragment store.
// ---------------------------------------------------------------------------
__global__ __launch_bounds__(256) void k_ln1_project(
    const float* __restrict__ z,
    const float* __restrict__ g1, const float* __restrict__ b1,
    const unsigned short* __restrict__ Wt,   // 6 matrices, (n,k) bf16
    const float* __restrict__ ba, const float* __restrict__ bga,
    const float* __restrict__ bb, const float* __restrict__ bgb,
    const float* __restrict__ bg,
    unsigned short* __restrict__ a_t,
    unsigned short* __restrict__ b_t,
    unsigned short* __restrict__ gate)
{
    __shared__ unsigned short sA[8][16 * CC];
    const int wave = threadIdx.x >> 5;
    const int lane = threadIdx.x & 31;
    const int lrow = lane & 15;
    const int hi   = lane >> 4;
    const int m0   = blockIdx.x * 128 + wave * 16;

    const float4 gv = *(const float4*)(g1 + lane * 4);
    const float4 bv = *(const float4*)(b1 + lane * 4);
    unsigned short* myA = sA[wave];

    for (int r = 0; r < 16; ++r) {
        const float4 x = *(const float4*)(z + (size_t)(m0 + r) * CC + lane * 4);
        const float mean = wred(x.x + x.y + x.z + x.w) * (1.0f / 128.0f);
        const float d0 = x.x - mean, d1 = x.y - mean, d2 = x.z - mean, d3 = x.w - mean;
        const float var = wred(d0 * d0 + d1 * d1 + d2 * d2 + d3 * d3) * (1.0f / 128.0f);
        const float inv = rsqrtf(var + 1e-5f);
        uint2 pk;
        pk.x = (unsigned)f2bf(d0 * inv * gv.x + bv.x) | ((unsigned)f2bf(d1 * inv * gv.y + bv.y) << 16);
        pk.y = (unsigned)f2bf(d2 * inv * gv.z + bv.z) | ((unsigned)f2bf(d3 * inv * gv.w + bv.w) << 16);
        *(uint2*)(myA + r * CC + lane * 4) = pk;
    }
    __syncthreads();

    const unsigned short* arow = myA + lrow * CC;
    v16bf af[4];
#pragma unroll
    for (int kk = 0; kk < 4; ++kk) af[kk] = load_afrag(arow, kk * 32, hi);

#pragma unroll 1
    for (int nt = 0; nt < 8; ++nt) {
        const int n = nt * 16 + lrow;
        v8f accA = {}, accGA = {}, accB = {}, accGB = {}, accG = {};
#pragma unroll
        for (int kk = 0; kk < 4; ++kk) {
            const int k0 = kk * 32;
            const v16bf fA  = load_bfrag(Wt + (size_t)(0 * CC + n) * CC, k0, hi);
            const v16bf fGA = load_bfrag(Wt + (size_t)(1 * CC + n) * CC, k0, hi);
            const v16bf fB  = load_bfrag(Wt + (size_t)(2 * CC + n) * CC, k0, hi);
            const v16bf fGB = load_bfrag(Wt + (size_t)(3 * CC + n) * CC, k0, hi);
            const v16bf fG  = load_bfrag(Wt + (size_t)(4 * CC + n) * CC, k0, hi);
            accA  = WMMA_BF16(af[kk], fA,  accA);
            accGA = WMMA_BF16(af[kk], fGA, accGA);
            accB  = WMMA_BF16(af[kk], fB,  accB);
            accGB = WMMA_BF16(af[kk], fGB, accGB);
            accG  = WMMA_BF16(af[kk], fG,  accG);
        }
        const float vba = ba[n], vbga = bga[n], vbb = bb[n], vbgb = bgb[n], vbg = bg[n];
        unsigned short ha[8], hb[8];
#pragma unroll
        for (int r = 0; r < 8; ++r) {
            ha[r] = f2bf((accA[r] + vba) * sigmoidf(accGA[r] + vbga));
            hb[r] = f2bf((accB[r] + vbb) * sigmoidf(accGB[r] + vbgb));
        }
        // transposed (channel-major) stores: lane covers M = hi*8 .. hi*8+7
        *(uint4*)(a_t + (size_t)n * MTOT + m0 + hi * 8) = pack8(ha);
        *(uint4*)(b_t + (size_t)n * MTOT + m0 + hi * 8) = pack8(hb);
#pragma unroll
        for (int r = 0; r < 8; ++r) {
            const int m = m0 + r + 8 * hi;
            gate[(size_t)m * CC + n] = f2bf(sigmoidf(accG[r] + vbg));
        }
    }
}

// ---------------------------------------------------------------------------
// Triangle einsum: per channel c, O_c(384x384) = A_c @ B_c^T, K = 384.
// Block = 8 waves -> 128(i) x 128(j) tile. The full 128x384 B panel (96 KB,
// contiguous in channel-major layout) is staged once into LDS with async
// global->LDS copies; each wave then runs 8 j-chains sharing one A fragment
// per K step (8 WMMAs per A load -> ~64 FLOP/B from L2).
// ---------------------------------------------------------------------------
__global__ __launch_bounds__(256) void k_triangle(
    const unsigned short* __restrict__ a_t,
    const unsigned short* __restrict__ b_t,
    float* __restrict__ o)
{
    __shared__ unsigned short sB[128 * NN];   // 96 KB
    const int tid  = threadIdx.x;
    const int wave = tid >> 5, lane = tid & 31;
    const int lrow = lane & 15, hi = lane >> 4;
    const int c  = blockIdx.z;
    const int i0 = blockIdx.y * 128;
    const int j0 = blockIdx.x * 128;

    // Stage B panel: rows j0..j0+127 are contiguous (flat 96 KB copy).
    const unsigned short* Bsrc = b_t + (size_t)c * MTOT + (size_t)j0 * NN;
#pragma unroll
    for (int it = 0; it < (128 * NN) / (8 * 256); ++it) {
        const int idx = (tid + it * 256) * 8;          // 16B chunks
        async_copy16(Bsrc + idx, sB + idx);
    }
    async_wait0();
    __syncthreads();

    const unsigned short* arow = a_t + (size_t)c * MTOT + (size_t)(i0 + wave * 16 + lrow) * NN;

    v8f acc[8];
#pragma unroll
    for (int nt = 0; nt < 8; ++nt) acc[nt] = (v8f){};

#pragma unroll 1
    for (int kk = 0; kk < 12; ++kk) {
        const int k0 = kk * 32;
        if (kk + 1 < 12) __builtin_prefetch(arow + k0 + 32, 0, 0);  // global_prefetch_b8
        const v16bf fa = load_afrag(arow, k0, hi);
#pragma unroll
        for (int nt = 0; nt < 8; ++nt) {
            const v16bf fb = load_bfrag(sB + (nt * 16 + lrow) * NN, k0, hi);
            acc[nt] = WMMA_BF16(fa, fb, acc[nt]);
        }
    }

#pragma unroll
    for (int nt = 0; nt < 8; ++nt) {
        const int j = j0 + nt * 16 + lrow;
#pragma unroll
        for (int r = 0; r < 8; ++r) {
            const int i = i0 + wave * 16 + r + 8 * hi;
            o[((size_t)i * NN + j) * CC + c] = acc[nt][r];
        }
    }
}

// ---------------------------------------------------------------------------
// LN2 over o, project with Wo, add bo, multiply by gate -> fp32 out.
// ---------------------------------------------------------------------------
__global__ __launch_bounds__(256) void k_ln2_out(
    const float* __restrict__ o,
    const float* __restrict__ g2, const float* __restrict__ b2,
    const unsigned short* __restrict__ Wot,   // (n,k) bf16
    const float* __restrict__ bo,
    const unsigned short* __restrict__ gate,
    float* __restrict__ out)
{
    __shared__ unsigned short sA[8][16 * CC];
    const int wave = threadIdx.x >> 5;
    const int lane = threadIdx.x & 31;
    const int lrow = lane & 15;
    const int hi   = lane >> 4;
    const int m0   = blockIdx.x * 128 + wave * 16;

    const float4 gv = *(const float4*)(g2 + lane * 4);
    const float4 bv = *(const float4*)(b2 + lane * 4);
    unsigned short* myA = sA[wave];

    for (int r = 0; r < 16; ++r) {
        const float4 x = *(const float4*)(o + (size_t)(m0 + r) * CC + lane * 4);
        const float mean = wred(x.x + x.y + x.z + x.w) * (1.0f / 128.0f);
        const float d0 = x.x - mean, d1 = x.y - mean, d2 = x.z - mean, d3 = x.w - mean;
        const float var = wred(d0 * d0 + d1 * d1 + d2 * d2 + d3 * d3) * (1.0f / 128.0f);
        const float inv = rsqrtf(var + 1e-5f);
        uint2 pk;
        pk.x = (unsigned)f2bf(d0 * inv * gv.x + bv.x) | ((unsigned)f2bf(d1 * inv * gv.y + bv.y) << 16);
        pk.y = (unsigned)f2bf(d2 * inv * gv.z + bv.z) | ((unsigned)f2bf(d3 * inv * gv.w + bv.w) << 16);
        *(uint2*)(myA + r * CC + lane * 4) = pk;
    }
    __syncthreads();

    const unsigned short* arow = myA + lrow * CC;
    v16bf af[4];
#pragma unroll
    for (int kk = 0; kk < 4; ++kk) af[kk] = load_afrag(arow, kk * 32, hi);

#pragma unroll 1
    for (int nt = 0; nt < 8; ++nt) {
        const int n = nt * 16 + lrow;
        v8f acc = {};
#pragma unroll
        for (int kk = 0; kk < 4; ++kk)
            acc = WMMA_BF16(af[kk], load_bfrag(Wot + (size_t)n * CC, kk * 32, hi), acc);
        const float vbo = bo[n];
#pragma unroll
        for (int r = 0; r < 8; ++r) {
            const int m = m0 + r + 8 * hi;
            out[(size_t)m * CC + n] = bf2f(gate[(size_t)m * CC + n]) * (acc[r] + vbo);
        }
    }
}

// ---------------------------------------------------------------------------
extern "C" void kernel_launch(void* const* d_in, const int* in_sizes, int n_in,
                              void* d_out, int out_size, void* d_ws, size_t ws_size,
                              hipStream_t stream) {
    (void)in_sizes; (void)n_in; (void)out_size; (void)ws_size;
    const float* z   = (const float*)d_in[0];
    const float* g1  = (const float*)d_in[1];
    const float* b1  = (const float*)d_in[2];
    const float* g2  = (const float*)d_in[3];
    const float* b2  = (const float*)d_in[4];
    const float* Wa  = (const float*)d_in[5];
    const float* ba  = (const float*)d_in[6];
    const float* Wga = (const float*)d_in[7];
    const float* bga = (const float*)d_in[8];
    const float* Wb  = (const float*)d_in[9];
    const float* bb  = (const float*)d_in[10];
    const float* Wgb = (const float*)d_in[11];
    const float* bgb = (const float*)d_in[12];
    const float* Wg  = (const float*)d_in[13];
    const float* bg  = (const float*)d_in[14];
    const float* Wo  = (const float*)d_in[15];
    const float* bo  = (const float*)d_in[16];
    float* out = (float*)d_out;

    char* ws = (char*)d_ws;
    unsigned short* Wt   = (unsigned short*)ws; ws += (size_t)6 * CC * CC * 2;    // 192 KB
    unsigned short* a_t  = (unsigned short*)ws; ws += (size_t)CC * MTOT * 2;      // 37.75 MB
    unsigned short* b_t  = (unsigned short*)ws; ws += (size_t)CC * MTOT * 2;      // 37.75 MB
    unsigned short* gate = (unsigned short*)ws; ws += (size_t)MTOT * CC * 2;      // 37.75 MB
    float*          obuf = (float*)ws;                                            // 75.5 MB

    const float* Wsrc[6] = {Wa, Wga, Wb, Wgb, Wg, Wo};
    for (int m = 0; m < 6; ++m)
        k_wprep<<<128, 128, 0, stream>>>(Wsrc[m], Wt + (size_t)m * CC * CC);

    k_ln1_project<<<MTOT / 128, 256, 0, stream>>>(z, g1, b1, Wt, ba, bga, bb, bgb, bg,
                                                  a_t, b_t, gate);
    k_triangle<<<dim3(NN / 128, NN / 128, CC), 256, 0, stream>>>(a_t, b_t, obuf);
    k_ln2_out<<<MTOT / 128, 256, 0, stream>>>(obuf, g2, b2, Wt + (size_t)5 * CC * CC,
                                              bo, gate, out);
}
